// Net_53781580480722
// MI455X (gfx1250) — matrix operand order repaired and checked by reading, not compiled, not run on hardware
//
#include <hip/hip_runtime.h>
#include <hip/hip_bf16.h>

// ---------------------------------------------------------------------------
// Fused persistent LSTM scan for MI455X (gfx1250, wave32, WMMA bf16 + TDM).
//   B=512, T=1024, HS=128, F=24 (23 points + time).
// One workgroup = 16 batch rows (WMMA M), 8 waves; wave w owns hidden cols
// [16w,16w+16) and the matching i/f/g/o gate tiles. U/W(+bias row) live in
// VGPRs as pre-packed bf16 B-fragments; h lives in 4KB LDS between steps;
// the 128->2 sigmoid head is fused. The per-step 16x23 fp32 x-tile is
// double-buffered in LDS and prefetched by the Tensor Data Mover
// (tensor_load_to_lds, synced with s_wait_tensorcnt) one step ahead.
// Nonlinearities use v_exp_f32 + v_rcp_f32 only (no IEEE div expansion).
// ---------------------------------------------------------------------------

#define HSIZE 128
#define NINP  23
#define TLEN  1024
#define BATCH 512
#define ROWS  16   // batch rows per workgroup (WMMA M)
#define NWAVE 8    // 8 waves * 16 cols = 128 hidden columns

typedef __attribute__((ext_vector_type(16))) __bf16 v16bf;
typedef __attribute__((ext_vector_type(8)))  __bf16 v8bf;
typedef __attribute__((ext_vector_type(8)))  float  v8f;

typedef __attribute__((ext_vector_type(4))) unsigned int u32x4;
typedef __attribute__((ext_vector_type(8))) int          i32x8;
typedef __attribute__((ext_vector_type(4))) int          i32x4;

#if __has_builtin(__builtin_amdgcn_tensor_load_to_lds) && \
    __has_builtin(__builtin_amdgcn_s_wait_tensorcnt)
#define USE_TDM 1
#else
#define USE_TDM 0
#endif

__device__ __forceinline__ float fast_rcp(float x) {
#if __has_builtin(__builtin_amdgcn_rcpf)
    return __builtin_amdgcn_rcpf(x);      // single v_rcp_f32
#else
    return 1.0f / x;
#endif
}
__device__ __forceinline__ float sigmoid_f(float x) {
    return fast_rcp(1.0f + __expf(-x));
}
__device__ __forceinline__ float tanh_f(float x) {
    // tanh(x) = 1 - 2/(1+exp(2x)); saturates correctly for |x| large
    return 1.0f - 2.0f * fast_rcp(1.0f + __expf(2.0f * x));
}

#if USE_TDM
// Issue a TDM load of a 2D fp32 tile (tile_dim1=16 rows x tile_dim0=23 cols,
// row stride TLEN*NINP elements) from `gaddr` into LDS offset `lds_off`.
// This toolchain (clang-23 / therock-10.0 headers) uses the 6-arg builtin:
//   (u32x4 g0, i32x8 g1, i32x4 g2, i32x4 g3, i32x8 g4, i32 cpol)
__device__ __forceinline__ void tdm_load_xtile(unsigned long long gaddr,
                                               unsigned lds_off) {
    // ---- D# group 0 (128b): count=1, lds_addr, 57b global_addr, type=2 ----
    u32x4 g0;
    g0[0] = 1u;                                   // count=1, user mode
    g0[1] = lds_off;                              // lds_addr[31:0]
    g0[2] = (unsigned)(gaddr & 0xFFFFFFFFu);      // global_addr[31:0]
    g0[3] = (unsigned)((gaddr >> 32) & 0x1FFFFFFu) | (2u << 30); // [56:32]|type=2
    // ---- D# group 1 (256b) ----
    i32x8 g1;
    g1[0] = (int)(2u << 16);                      // wg_mask=0, data_size=2 (4B)
    g1[1] = (int)((unsigned)NINP << 16);          // tensor_dim0[15:0] in [31:16]
    g1[2] = (int)((unsigned)ROWS << 16);          // tensor_dim0 hi=0 | tensor_dim1 lo
    g1[3] = (int)((unsigned)NINP << 16);          // tensor_dim1 hi=0 | tile_dim0=23
    g1[4] = (int)(unsigned)ROWS;                  // tile_dim1=16, tile_dim2=0
    g1[5] = (int)(unsigned)(TLEN * NINP);         // tensor_dim0_stride lo32
    g1[6] = 0;                                    // stride hi16 | dim1_stride lo16
    g1[7] = 0;
    i32x4 gz4 = {0, 0, 0, 0};                     // groups 2/3 unused (2D tile)
    i32x8 gz8 = {0, 0, 0, 0, 0, 0, 0, 0};
    __builtin_amdgcn_tensor_load_to_lds(g0, g1, gz4, gz4, gz8, 0);
}
#endif

__global__ __launch_bounds__(256, 1)
void lstm_fused_scan(const float* __restrict__ points,
                     const float* __restrict__ times,
                     const float* __restrict__ W,     // [24, 512]
                     const float* __restrict__ U,     // [128, 512]
                     const float* __restrict__ bias,  // [512]
                     const float* __restrict__ Wc,    // [128, 2]
                     const float* __restrict__ bc,    // [2]
                     float* __restrict__ out)         // [512, 1024, 2]
{
    __shared__ __bf16 h_lds[ROWS * HSIZE];          // 4 KB: current hidden state
    __shared__ float  out_part[NWAVE][ROWS][2];     // head partial sums
    __shared__ float  x_stage[2][ROWS * NINP];      // double-buffered x tile

    const int tid   = threadIdx.x;
    const int wave  = tid >> 5;        // 0..7
    const int lane  = tid & 31;
    const int lhalf = lane >> 4;       // 0 or 1 (lane half per WMMA layout)
    const int l16   = lane & 15;
    const int b0    = blockIdx.x * ROWS;

    // ---- Prologue: pack U and W(+bias as row 24) into bf16 B-fragments ----
    // B-fragment element e of lane: row K = kbase + 8*lhalf + e + (e>=8 ? 8 : 0), col = base+l16.
    v16bf ufrag[4][4];   // [k-slice][gate]  : 128 VGPRs
    v16bf wfrag[4];      // [gate]; K: 0..22 = W rows, 23 = W time row,
                         //          24 = bias (x K=24 is constant 1), 25..31 = 0
    #pragma unroll
    for (int g = 0; g < 4; ++g) {
        const int col = g * HSIZE + wave * 16 + l16;
        #pragma unroll
        for (int kt = 0; kt < 4; ++kt) {
            #pragma unroll
            for (int e = 0; e < 16; ++e) {
                const int k = kt * 32 + lhalf * 8 + e + ((e >= 8) ? 8 : 0);
                ufrag[kt][g][e] = (__bf16)U[k * (4 * HSIZE) + col];
            }
        }
        #pragma unroll
        for (int e = 0; e < 16; ++e) {
            const int k = lhalf * 8 + e + ((e >= 8) ? 8 : 0);
            float wv = 0.0f;
            if (k < NINP + 1)       wv = W[k * (4 * HSIZE) + col];
            else if (k == NINP + 1) wv = bias[col];   // bias via constant-1 feature
            wfrag[g][e] = (__bf16)wv;
        }
    }

    // Head weights: lane (m=l16, c=lhalf) needs Wc[16*wave + jj][lhalf].
    float wcreg[16];
    #pragma unroll
    for (int jj = 0; jj < 16; ++jj)
        wcreg[jj] = Wc[(wave * 16 + jj) * 2 + lhalf];
    const float bc_l = bc[lhalf];

    // ---- Initial state: h = c = 0; prefetch x tile for t=0 ----
    v8f c_state;
    #pragma unroll
    for (int v = 0; v < 8; ++v) c_state[v] = 0.0f;
    for (int i = tid; i < ROWS * HSIZE; i += 256) h_lds[i] = (__bf16)0.0f;

    const unsigned long long pbase =
        (unsigned long long)(uintptr_t)(points) +
        (unsigned long long)b0 * TLEN * NINP * 4ull;
#if USE_TDM
    const unsigned xstage_lds0 =
        (unsigned)(uintptr_t)(void*)&x_stage[0][0];
    const unsigned xstage_lds1 =
        (unsigned)(uintptr_t)(void*)&x_stage[1][0];
    if (wave == 0) {
        tdm_load_xtile(pbase, xstage_lds0);               // tile for t=0
        __builtin_amdgcn_s_wait_tensorcnt(0);
    }
#else
    for (int i = tid; i < ROWS * NINP; i += 256)
        x_stage[0][i] = points[(size_t)(b0 + i / NINP) * TLEN * NINP + i % NINP];
#endif
    __syncthreads();

    const float* trow_base = times + (size_t)(b0 + l16) * TLEN;

    for (int t = 0; t < TLEN; ++t) {
        const int p = t & 1;

#if USE_TDM
        // Prefetch next step's x tile into the other buffer (DMA overlaps
        // the whole WMMA chain below; completion waited before last barrier).
        if (wave == 0 && (t + 1) < TLEN) {
            tdm_load_xtile(pbase + (unsigned long long)(t + 1) * NINP * 4ull,
                           p ? xstage_lds0 : xstage_lds1);
        }
#endif

        // ---- A-fragment of x = [points | time | 1], K padded 24->32 ----
        // elem e -> K = lhalf*8 + e + (e>=8 ? 8 : 0); row m = l16.
        const float* xs   = &x_stage[p][l16 * NINP];
        const float  tval = trow_base[t];
        v16bf ax;
        #pragma unroll
        for (int e = 0; e < 16; ++e) {
            const int k = lhalf * 8 + e + ((e >= 8) ? 8 : 0);
            float v;
            if (k < NINP)            v = xs[k];
            else if (k == NINP)      v = tval;
            else if (k == NINP + 1)  v = 1.0f;   // constant feature -> bias
            else                     v = 0.0f;
            ax[e] = (__bf16)v;
        }

        // ---- gates = [x|1]@[W;bias] + h@U   (acc starts at inline 0) ----
        v8f acc[4];
        const v8f zero8 = {};
        #pragma unroll
        for (int g = 0; g < 4; ++g)
            acc[g] = __builtin_amdgcn_wmma_f32_16x16x32_bf16(
                false, ax, false, wfrag[g], (short)0, zero8, false, false);

        #pragma unroll
        for (int kt = 0; kt < 4; ++kt) {
            // A-fragment of h from LDS: two 16B ds loads per lane.
            union { v16bf v; v8bf h[2]; } ah;
            const int rbase = l16 * HSIZE + kt * 32 + lhalf * 8;
            ah.h[0] = *(const v8bf*)&h_lds[rbase];
            ah.h[1] = *(const v8bf*)&h_lds[rbase + 16];
            #pragma unroll
            for (int g = 0; g < 4; ++g)
                acc[g] = __builtin_amdgcn_wmma_f32_16x16x32_bf16(
                    false, ah.v, false, ufrag[kt][g], (short)0, acc[g],
                    false, false);
        }

        // ---- Nonlinearities + state update (wave-local: i/f/g/o aligned) ----
        v8f hnew;
        #pragma unroll
        for (int v = 0; v < 8; ++v) {
            const float ig = sigmoid_f(acc[0][v]);
            const float fg = sigmoid_f(acc[1][v]);
            const float gg = tanh_f(acc[2][v]);
            const float og = sigmoid_f(acc[3][v]);
            const float cv = fg * c_state[v] + ig * gg;
            c_state[v] = cv;
            hnew[v] = og * tanh_f(cv);
        }

        __syncthreads();   // all waves done reading h_lds / wave0 done out_part
        // C/D layout: elem v -> row m = v + 8*lhalf, col = 16*wave + l16.
        #pragma unroll
        for (int v = 0; v < 8; ++v) {
            const int m = v + 8 * lhalf;
            h_lds[m * HSIZE + wave * 16 + l16] = (__bf16)hnew[v];
        }
        __syncthreads();   // new h visible

        // ---- Fused head: each wave contributes a 16-wide partial dot ----
        {
            const int m = l16, c = lhalf;
            float ph = 0.0f;
            #pragma unroll
            for (int jj = 0; jj < 16; ++jj)
                ph += (float)h_lds[m * HSIZE + wave * 16 + jj] * wcreg[jj];
            out_part[wave][m][c] = ph;
        }

#if USE_TDM
        if (wave == 0) __builtin_amdgcn_s_wait_tensorcnt(0);  // t+1 tile ready
#else
        __syncthreads();
        {
            const int pn = (t + 1) & 1;
            if ((t + 1) < TLEN)
                for (int i = tid; i < ROWS * NINP; i += 256)
                    x_stage[pn][i] = points[(size_t)(b0 + i / NINP) * TLEN * NINP
                                            + (size_t)(t + 1) * NINP + i % NINP];
        }
#endif
        __syncthreads();   // partials (and next x tile) visible

        if (wave == 0) {
            const int m = l16, c = lhalf;
            float s = bc_l;
            #pragma unroll
            for (int w2 = 0; w2 < NWAVE; ++w2) s += out_part[w2][m][c];
            out[(((size_t)(b0 + m)) * TLEN + t) * 2 + c] = sigmoid_f(s);
        }
    }
}

extern "C" void kernel_launch(void* const* d_in, const int* in_sizes, int n_in,
                              void* d_out, int out_size, void* d_ws, size_t ws_size,
                              hipStream_t stream) {
    (void)in_sizes; (void)n_in; (void)d_ws; (void)ws_size; (void)out_size;
    const float* points = (const float*)d_in[0];
    const float* times  = (const float*)d_in[1];
    const float* W      = (const float*)d_in[2];
    const float* U      = (const float*)d_in[3];
    const float* bias   = (const float*)d_in[4];
    const float* Wc     = (const float*)d_in[5];
    const float* bc     = (const float*)d_in[6];
    float* out = (float*)d_out;

    dim3 grid(BATCH / ROWS);   // 32 workgroups
    dim3 block(256);           // 8 waves (wave32)
    lstm_fused_scan<<<grid, block, 0, stream>>>(points, times, W, U, bias, Wc,
                                                bc, out);
}
